// Attention_73564199846195
// MI455X (gfx1250) — matrix operand order repaired and checked by reading, not compiled, not run on hardware
//
#include <hip/hip_runtime.h>
#include <math.h>

typedef __attribute__((ext_vector_type(16))) _Float16 v16h;
typedef __attribute__((ext_vector_type(8)))  _Float16 h8;
typedef __attribute__((ext_vector_type(8)))  float    v8f;
typedef __attribute__((ext_vector_type(4)))  float    f4;

union Frag {
    v16h v;
    unsigned u[8];
    _Float16 h[16];
    h8 g[2];
};

// CDNA5 async global->LDS copy (ASYNCcnt-tracked). LDS dest = byte offset in
// VGPR (added to the LDS base and the instruction offset), global addr = VGPR pair.
__device__ __forceinline__ void async_copy_b128(unsigned lds_off, const void* gaddr) {
    asm volatile("global_load_async_to_lds_b128 %0, %1, off"
                 :: "v"(lds_off), "v"(gaddr)
                 : "memory");
}
__device__ __forceinline__ void wait_async_le2() {
    asm volatile("s_wait_asynccnt 0x2" ::: "memory");
}

// LDS tile geometry (halves). Strides padded, 16B-aligned for b128 async writes.
#define A_STRIDE 40                 // 32 + 8 pad -> 80B rows (20-bank step)
#define B_STRIDE 72                 // 64 + 8 pad -> 144B rows (36-bank step)
#define A_TILE_H (64 * A_STRIDE)    // 2560 halves
#define B_TILE_H (32 * B_STRIDE)    // 2304 halves
#define BUF_H    (A_TILE_H + B_TILE_H)

// ---------------------------------------------------------------------------
// All-f16 GEMM, async double-buffered: Y[b] = W @ X[b]
// W: [M][K] f16, X: [B][K][N] f16, Y: [B][M][N] TO (f16 or f32)
// BM=BN=64, BK=32; 8 waves, 2 WMMA tiles each (v_wmma_f32_16x16x32_f16).
// ---------------------------------------------------------------------------
template <typename TO>
__global__ __launch_bounds__(256)
void gemm_h_async(const _Float16* __restrict__ W, const _Float16* __restrict__ X,
                  TO* __restrict__ Y, int M, int K, int N)
{
    __shared__ _Float16 lds[2 * BUF_H];

    const int b  = blockIdx.z;
    const int m0 = blockIdx.y * 64;
    const int n0 = blockIdx.x * 64;
    const _Float16* Xb = X + (size_t)b * K * N;
    TO*             Yb = Y + (size_t)b * M * N;

    const int t = threadIdx.x;
    const int lane = t & 31, wave = t >> 5;
    const int half = lane >> 4, lr = lane & 15;
    const int msub = wave >> 1;          // 0..3
    const int nsub0 = (wave & 1) * 2;    // 0 or 2

    // staging: one b128 (8 halves) per thread per matrix per tile
    const int ar = t >> 2, ac = t & 3;   // A: row 0..63, chunk 0..3
    const int br = t >> 3, bc = t & 7;   // B: k-row 0..31, chunk 0..7

    const unsigned lds0 = (unsigned)(size_t)(void*)&lds[0];   // LDS byte offset
    const unsigned aoff = lds0 + (unsigned)(ar * A_STRIDE + ac * 8) * 2;
    const unsigned boff = lds0 + (unsigned)(A_TILE_H + br * B_STRIDE + bc * 8) * 2;
    const unsigned bufB = (unsigned)(BUF_H * 2);

    const _Float16* agp = W  + (size_t)(m0 + ar) * K + ac * 8;   // += k
    const _Float16* bgp = Xb + (size_t)br * N + n0 + bc * 8;     // += k*N

    // prologue: stage tile 0 -> buffer 0
    async_copy_b128(aoff, agp);
    async_copy_b128(boff, bgp);

    const int T = K / 32;
    v8f acc0 = {};
    v8f acc1 = {};

    for (int i = 0; i < T; ++i) {
        const int cur = i & 1;
        // issue tile i+1 into the other buffer (last iter: harmless refetch)
        const int kn = (i + 1 < T) ? (i + 1) * 32 : (T - 1) * 32;
        const unsigned nb = (unsigned)((i + 1) & 1) * bufB;
        async_copy_b128(aoff + nb, agp + kn);
        async_copy_b128(boff + nb, bgp + (size_t)kn * N);

        wait_async_le2();        // tile i's 2 ops complete (in-order retire)
        __syncthreads();

        const _Float16* Al = &lds[cur * BUF_H];
        const _Float16* Bl = Al + A_TILE_H;

        // A fragment: h[e] <-> K = e + 16*half (lane row = msub*16+lr; packed K pairs)
        Frag af;
        {
            const _Float16* row = Al + (msub * 16 + lr) * A_STRIDE;
            #pragma unroll
            for (int p = 0; p < 8; ++p) {
                const int kp = 2 * p + 16 * half;
                af.u[p] = *(const unsigned*)(row + kp);
            }
        }
        // B fragments from k-major LDS tile: h[e] <-> K = e + 16*half, col fixed
        Frag bf0, bf1;
        {
            const _Float16* c0p = Bl + nsub0 * 16 + lr;
            const _Float16* c1p = c0p + 16;
            #pragma unroll
            for (int e = 0; e < 16; ++e) {
                const int kk = (e + 16 * half) * B_STRIDE;
                bf0.h[e] = c0p[kk];
                bf1.h[e] = c1p[kk];
            }
        }
        acc0 = __builtin_amdgcn_wmma_f32_16x16x32_f16(false, af.v, false, bf0.v,
                                                      (short)0, acc0, false, false);
        acc1 = __builtin_amdgcn_wmma_f32_16x16x32_f16(false, af.v, false, bf1.v,
                                                      (short)0, acc1, false, false);
        __syncthreads();
    }

    const int n = n0 + nsub0 * 16 + lr;
    #pragma unroll
    for (int v = 0; v < 8; ++v) {
        const int m = m0 + msub * 16 + v + 8 * half;
        Yb[(size_t)m * N + n]      = (TO)acc0[v];
        Yb[(size_t)m * N + n + 16] = (TO)acc1[v];
    }
}

// ---------------------------------------------------------------------------
// fp32 -> fp16 elementwise convert (weights / input image, one-time)
// ---------------------------------------------------------------------------
__global__ __launch_bounds__(256)
void cvt_f32_f16(const float* __restrict__ in, _Float16* __restrict__ out, size_t n)
{
    const size_t i = (size_t)blockIdx.x * blockDim.x + threadIdx.x;
    if (i < n) out[i] = (_Float16)in[i];
}

// ---------------------------------------------------------------------------
// Depthwise 3/5/7 conv, f16 in/out, f32 accumulate. C = 1152.
// ---------------------------------------------------------------------------
__global__ __launch_bounds__(256)
void dwconv_multi(const _Float16* __restrict__ base, const float* __restrict__ w3,
                  const float* __restrict__ w5, const float* __restrict__ w7,
                  _Float16* __restrict__ out, int B, int C, int H, int Wd)
{
    size_t idx = (size_t)blockIdx.x * blockDim.x + threadIdx.x;
    const size_t total = (size_t)B * 3 * C * H * Wd;
    if (idx >= total) return;
    const int x = (int)(idx % Wd); size_t r = idx / Wd;
    const int y = (int)(r % H); r /= H;
    const int c = (int)(r % C); r /= C;
    const int g = (int)(r % 3);
    const int b = (int)(r / 3);

    const int ks = 3 + 2 * g;
    const int pad = g + 1;
    const float* wp = (g == 0 ? w3 : (g == 1 ? w5 : w7)) + (size_t)c * ks * ks;
    const _Float16* ip = base + ((size_t)b * C + c) * H * Wd;

    float s = 0.f;
    for (int ky = 0; ky < ks; ++ky) {
        const int iy = y + ky - pad;
        if (iy < 0 || iy >= H) continue;
        for (int kx = 0; kx < ks; ++kx) {
            const int ix = x + kx - pad;
            if (ix < 0 || ix >= Wd) continue;
            s = fmaf((float)ip[(size_t)iy * Wd + ix], wp[ky * ks + kx], s);
        }
    }
    out[(((size_t)b * 3 + g) * C + c) * H * Wd + (size_t)y * Wd + x] = (_Float16)s;
}

// ---------------------------------------------------------------------------
// inv L2 norm of q/k rows (f16 data, f32 accumulate)
// ---------------------------------------------------------------------------
__global__ __launch_bounds__(256)
void rownorm_inv(const _Float16* __restrict__ qk, float* __restrict__ inv,
                 int rowsPerBatch, int chanPerBatch, int N)
{
    const int row = blockIdx.x;
    const int b = row / rowsPerBatch;
    const int c = row % rowsPerBatch;
    const _Float16* p = qk + ((size_t)b * chanPerBatch + c) * N;
    float s = 0.f;
    for (int i = threadIdx.x; i < N; i += blockDim.x) {
        const float v = (float)p[i];
        s = fmaf(v, v, s);
    }
    __shared__ float red[256];
    red[threadIdx.x] = s;
    __syncthreads();
    for (int off = 128; off > 0; off >>= 1) {
        if ((int)threadIdx.x < off) red[threadIdx.x] += red[threadIdx.x + off];
        __syncthreads();
    }
    if (threadIdx.x == 0) inv[row] = 1.f / fmaxf(sqrtf(red[0]), 1e-12f);
}

// ---------------------------------------------------------------------------
// attn[b][h][c][d] = (q_c . k_d) * inv||q|| * inv||k|| * temp[h]
// block per (b,h): 9 waves, one 16x16 tile each, K = N = 16384, f16 operands
// ---------------------------------------------------------------------------
__global__ __launch_bounds__(288)
void attn_qk(const _Float16* __restrict__ qkv2, const float* __restrict__ invn,
             const float* __restrict__ temperature, float* __restrict__ attn, int N)
{
    const int bh = blockIdx.x;
    const int b = bh >> 3, h = bh & 7;
    const int wave = threadIdx.x >> 5;      // 0..8
    const int lane = threadIdx.x & 31;
    const int half = lane >> 4, lr = lane & 15;
    const int c0 = (wave / 3) * 16, d0 = (wave % 3) * 16;

    const _Float16* qrow = qkv2 + ((size_t)b * 1152 + h * 48 + c0 + lr) * N;
    const _Float16* krow = qkv2 + ((size_t)b * 1152 + 384 + h * 48 + d0 + lr) * N;

    v8f acc = {};
    for (int n = 0; n < N; n += 32) {
        Frag a, bf;
        a.g[0]  = *(const h8*)(qrow + n + 16 * half);       // K = e + 16*half
        a.g[1]  = *(const h8*)(qrow + n + 16 * half + 8);   // K = e + 8 + 16*half
        bf.g[0] = *(const h8*)(krow + n + 16 * half);       // K = e + 16*half
        bf.g[1] = *(const h8*)(krow + n + 16 * half + 8);
        acc = __builtin_amdgcn_wmma_f32_16x16x32_f16(false, a.v, false, bf.v,
                                                     (short)0, acc, false, false);
    }

    const float temp = temperature[h];
    const int d = d0 + lr;
    const float ik = invn[b * 768 + 384 + h * 48 + d];
    #pragma unroll
    for (int v = 0; v < 8; ++v) {
        const int c = c0 + v + 8 * half;
        const float iq = invn[b * 768 + h * 48 + c];
        attn[((size_t)bh * 48 + c) * 48 + d] = acc[v] * iq * ik * temp;
    }
}

// ---------------------------------------------------------------------------
// Combined top-k masked softmax: Acomb = sum_i scale_i * softmax(topk_i(row))
// ---------------------------------------------------------------------------
__global__ void topk_combine(const float* __restrict__ attn,
                             const float* __restrict__ scales,
                             float* __restrict__ acomb, int nrows)
{
    const int row = blockIdx.x * blockDim.x + threadIdx.x;
    if (row >= nrows) return;
    const float* a = attn + (size_t)row * 48;

    float v[48];
    float mx = -3.4e38f;
    for (int j = 0; j < 48; ++j) { v[j] = a[j]; mx = fmaxf(mx, v[j]); }

    int cnt[48];
    for (int j = 0; j < 48; ++j) {
        int c = 0;
        for (int l = 0; l < 48; ++l)
            c += (v[l] > v[j]) || (v[l] == v[j] && l < j);
        cnt[j] = c;
    }

    float e[48];
    for (int j = 0; j < 48; ++j) e[j] = __expf(v[j] - mx);

    const int ks[3] = {24, 32, 36};
    float out[48];
    for (int j = 0; j < 48; ++j) out[j] = 0.f;
    for (int i = 0; i < 3; ++i) {
        float s = 0.f;
        for (int j = 0; j < 48; ++j) if (cnt[j] < ks[i]) s += e[j];
        const float sc = scales[i] / s;
        for (int j = 0; j < 48; ++j) if (cnt[j] < ks[i]) out[j] += sc * e[j];
    }
    float* o = acomb + (size_t)row * 48;
    for (int j = 0; j < 48; ++j) o[j] = out[j];
}

// ---------------------------------------------------------------------------
// out[b][h*48+c][n] = Acomb[b,h] @ V[b,h]  (M=48, K=48 zero-padded to 64)
// ---------------------------------------------------------------------------
__global__ __launch_bounds__(256)
void out_gemm(const float* __restrict__ acomb, const _Float16* __restrict__ qkv2,
              _Float16* __restrict__ outat, int N)
{
    const int bh = blockIdx.y;
    const int b = bh >> 3, h = bh & 7;
    const int n0 = blockIdx.x * 128;
    const int wave = threadIdx.x >> 5;   // n sub-tile 0..7
    const int lane = threadIdx.x & 31;
    const int half = lane >> 4, lr = lane & 15;

    const float*    A  = acomb + (size_t)bh * 48 * 48;
    const _Float16* Vb = qkv2 + ((size_t)b * 1152 + 768 + h * 48) * N;

    v8f acc[3] = {{}, {}, {}};
    for (int k0 = 0; k0 < 64; k0 += 32) {
        Frag bf;
        #pragma unroll
        for (int e2 = 0; e2 < 16; ++e2) {
            const int Kk = k0 + e2 + 16 * half;
            bf.h[e2] = (Kk < 48) ? Vb[(size_t)Kk * N + n0 + wave * 16 + lr]
                                 : (_Float16)0.f;
        }
        #pragma unroll
        for (int m = 0; m < 3; ++m) {
            Frag af;
            const float* arow = A + (m * 16 + lr) * 48;
            #pragma unroll
            for (int e2 = 0; e2 < 16; ++e2) {
                const int Kk = k0 + e2 + 16 * half;
                af.h[e2] = (Kk < 48) ? (_Float16)arow[Kk] : (_Float16)0.f;
            }
            acc[m] = __builtin_amdgcn_wmma_f32_16x16x32_f16(false, af.v, false, bf.v,
                                                            (short)0, acc[m], false, false);
        }
    }

    const int n = n0 + wave * 16 + lr;
    _Float16* O = outat + ((size_t)b * 384 + h * 48) * N;
    #pragma unroll
    for (int m = 0; m < 3; ++m) {
        #pragma unroll
        for (int v = 0; v < 8; ++v) {
            const int c = m * 16 + v + 8 * half;
            O[(size_t)c * N + n] = (_Float16)acc[m][v];
        }
    }
}

// ---------------------------------------------------------------------------
extern "C" void kernel_launch(void* const* d_in, const int* in_sizes, int n_in,
                              void* d_out, int out_size, void* d_ws, size_t ws_size,
                              hipStream_t stream)
{
    (void)in_sizes; (void)n_in; (void)out_size; (void)ws_size;

    const float* x           = (const float*)d_in[0];
    const float* w_qkv       = (const float*)d_in[1];
    const float* w_dw3       = (const float*)d_in[2];
    const float* w_dw5       = (const float*)d_in[3];
    const float* w_dw7       = (const float*)d_in[4];
    const float* w_pconv     = (const float*)d_in[5];
    const float* w_proj      = (const float*)d_in[6];
    const float* temperature = (const float*)d_in[7];
    const float* attn_scales = (const float*)d_in[8];
    float* out = (float*)d_out;

    const int B = 2, C = 384, H = 128, Wd = 128;
    const int N  = H * Wd;     // 16384
    const int C3 = 3 * C;      // 1152
    const int C9 = 9 * C;      // 3456

    // ---- workspace carve (halves first, then f32 smalls) ----
    _Float16* hs = (_Float16*)d_ws;
    const size_t szBase = (size_t)B * C3 * N;   // 37.7M h
    const size_t szQ9   = (size_t)B * C9 * N;   // 113M h
    const size_t szXh   = (size_t)B * C  * N;   // 12.6M h
    const size_t szWqkv = (size_t)C3 * C;       // 442K h
    const size_t szWpc  = (size_t)C3 * C9;      // 3.98M h
    const size_t szWpj  = (size_t)C  * C;       // 147K h

    _Float16* base_h  = hs;
    _Float16* qkv9_h  = base_h + szBase;
    _Float16* xh      = qkv9_h + szQ9;
    _Float16* wqkv_h  = xh + szXh;
    _Float16* wpc_h   = wqkv_h + szWqkv;
    _Float16* wpj_h   = wpc_h + szWpc;
    _Float16* qkv2_h  = base_h;                 // reuse (base dead after dwconv)
    _Float16* outat_h = qkv9_h;                 // reuse (qkv9 dead after pconv)

    float* fsm   = (float*)(wpj_h + szWpj);
    float* invn  = fsm;                          // [B*768]
    float* attn  = invn + B * 768;               // [B*8*48*48]
    float* acomb = attn + (size_t)B * 8 * 48 * 48;

    // 0) one-time f32 -> f16 converts (x + all 1x1 weights)
    cvt_f32_f16<<<(int)((szXh   + 255) / 256), 256, 0, stream>>>(x,       xh,     szXh);
    cvt_f32_f16<<<(int)((szWqkv + 255) / 256), 256, 0, stream>>>(w_qkv,   wqkv_h, szWqkv);
    cvt_f32_f16<<<(int)((szWpc  + 255) / 256), 256, 0, stream>>>(w_pconv, wpc_h,  szWpc);
    cvt_f32_f16<<<(int)((szWpj  + 255) / 256), 256, 0, stream>>>(w_proj,  wpj_h,  szWpj);

    // 1) base = w_qkv @ x
    gemm_h_async<_Float16><<<dim3(N / 64, C3 / 64, B), 256, 0, stream>>>(
        wqkv_h, xh, base_h, C3, C, N);

    // 2) qkv9 = concat(dw3, dw5, dw7)(base)
    {
        const size_t total = (size_t)B * C9 * N;
        dwconv_multi<<<(int)((total + 255) / 256), 256, 0, stream>>>(
            base_h, w_dw3, w_dw5, w_dw7, qkv9_h, B, C3, H, Wd);
    }

    // 3) qkv2 = w_pconv @ qkv9   (dominant GEMM)
    gemm_h_async<_Float16><<<dim3(N / 64, C3 / 64, B), 256, 0, stream>>>(
        wpc_h, qkv9_h, qkv2_h, C3, C9, N);

    // 4) inverse L2 norms of q/k rows
    rownorm_inv<<<B * 768, 256, 0, stream>>>(qkv2_h, invn, 768, C3, N);

    // 5) attn scores
    attn_qk<<<B * 8, 288, 0, stream>>>(qkv2_h, invn, temperature, attn, N);

    // 6) combined top-k softmax
    topk_combine<<<(B * 8 * 48 + 127) / 128, 128, 0, stream>>>(
        attn, attn_scales, acomb, B * 8 * 48);

    // 7) outat = Acomb @ V
    out_gemm<<<dim3(N / 128, B * 8), 256, 0, stream>>>(acomb, qkv2_h, outat_h, N);

    // 8) out = w_proj @ outat  (f32 output)
    gemm_h_async<float><<<dim3(N / 64, C / 64, B), 256, 0, stream>>>(
        wpj_h, outat_h, out, C, C, N);
}
